// WindowAttention_72928544686434
// MI455X (gfx1250) — compile-verified, matrix operand order checked by reading, not run on hardware
//
#include <hip/hip_runtime.h>
#include <hip/hip_bf16.h>
#include <cstdint>

typedef __attribute__((ext_vector_type(16))) _Float16     v16h;
typedef __attribute__((ext_vector_type(8)))  _Float16     v8h;
typedef __attribute__((ext_vector_type(8)))  float        v8f;
typedef __attribute__((ext_vector_type(4)))  unsigned int u32x4;
typedef __attribute__((ext_vector_type(8)))  unsigned int u32x8;

#define HEADS 12
#define NTOK  64
#define DH    32
#define CDIM  384
#define BWIN  1152
#define MROWS (BWIN * NTOK)   // 73728
#define KSTEPS (CDIM / 32)    // 12

// Load a 16x32 f16 A-style fragment (also used for B since B layout == B^T in A layout).
// Per ISA 7.12.2: lanes 0-15 hold row=lane, K = 0..7 (V0-3) and 16..23 (V4-7);
// lanes 16-31 hold row=lane-16, K = 8..15 and 24..31.  -> two 16B loads per lane.
__device__ __forceinline__ v16h load_frag_a(const _Float16* tile, int ld) {
  const int lane = threadIdx.x & 31;
  const int lr   = lane & 15;
  const int klo  = (lane >> 4) << 3;          // 0 or 8
  const _Float16* p = tile + (size_t)lr * ld + klo;
  v8h lo = *(const v8h*)(p);
  v8h hi = *(const v8h*)(p + 16);
  v16h r;
#pragma unroll
  for (int i = 0; i < 8; ++i) { r[i] = lo[i]; r[i + 8] = hi[i]; }
  return r;
}

__device__ __forceinline__ v8f wmma_f16(v16h a, v16h b, v8f c) {
  return __builtin_amdgcn_wmma_f32_16x16x32_f16(false, a, false, b, (short)0, c,
                                                false, false);
}

// TDM: DMA a 2D f16 tile (tile_d0 K-elems x tile_d1 rows) from global to LDS.
// D# layout per CDNA5 ISA (groups 0 and 1, 2D tensor,
// VADDR2/VADDR3 omitted). Scalar-issued, tracked by TENSORcnt.
__device__ __forceinline__ void tdm_load_2d(unsigned int lds_off, const void* gptr,
                                            unsigned int tile_d0, unsigned int tile_d1,
                                            unsigned int tensor_d0, unsigned int tensor_d1,
                                            unsigned int stride0) {
  const unsigned long long ga = (unsigned long long)(uintptr_t)gptr;
  u32x4 g0;
  g0[0] = 1u;                                              // count=1 (valid user D#)
  g0[1] = lds_off;                                         // lds_addr (bytes)
  g0[2] = (unsigned int)(ga & 0xFFFFFFFFu);                // global_addr[31:0]
  g0[3] = (unsigned int)((ga >> 32) & 0x1FFFFFFu)          // global_addr[56:32]
          | (2u << 30);                                    // type = 2 ("image")
  u32x8 g1;
  g1[0] = (1u << 16);                                      // data_size=1 -> 2 bytes
  g1[1] = (tensor_d0 & 0xFFFFu) << 16;                     // tensor_dim0[15:0]
  g1[2] = (tensor_d0 >> 16) | ((tensor_d1 & 0xFFFFu) << 16);
  g1[3] = (tensor_d1 >> 16) | (tile_d0 << 16);             // tile_dim0
  g1[4] = tile_d1;                                         // tile_dim1 (tile_dim2=0)
  g1[5] = stride0;                                         // tensor_dim0_stride[31:0]
  g1[6] = 0u;
  g1[7] = 0u;
  asm volatile("tensor_load_to_lds %0, %1" :: "s"(g0), "s"(g1) : "memory");
}

// ---------------- prep kernels ----------------

__global__ void k_cvt_f32_to_f16(const float* __restrict__ in,
                                 _Float16* __restrict__ out, int n) {
  for (int i = blockIdx.x * blockDim.x + threadIdx.x; i < n;
       i += gridDim.x * blockDim.x)
    out[i] = (_Float16)in[i];
}

// w (K x N, row-major f32) -> wt (N x K, row-major f16)
__global__ void k_transpose_w(const float* __restrict__ w,
                              _Float16* __restrict__ wt) {
  int i = blockIdx.x * blockDim.x + threadIdx.x;  // exactly 384*384
  int n = i / CDIM, k = i % CDIM;
  wt[(size_t)n * CDIM + k] = (_Float16)w[(size_t)k * CDIM + n];
}

// continuous-position-bias MLP over the 15x15 log-coord table -> tbv[225][H]
__global__ void k_cpb_mlp(const float* __restrict__ w1, const float* __restrict__ b1,
                          const float* __restrict__ w2, const float* __restrict__ b2,
                          float* __restrict__ tbv) {
  int id = blockIdx.x * blockDim.x + threadIdx.x;
  if (id >= 225 * HEADS) return;
  int e = id / HEADS, h = id % HEADS;
  int i = e / 15, j = e % 15;
  float g0 = (float)(j - 7) / 7.0f * 8.0f;   // t[i,j,0] = f(rh[j])
  float g1 = (float)(i - 7) / 7.0f * 8.0f;   // t[i,j,1] = f(rw[i])
  float s0 = (g0 > 0.f) ? 1.f : ((g0 < 0.f) ? -1.f : 0.f);
  float s1 = (g1 > 0.f) ? 1.f : ((g1 < 0.f) ? -1.f : 0.f);
  float t0 = s0 * log2f(fabsf(g0) + 1.0f) / 3.0f;   // log2(8)=3
  float t1 = s1 * log2f(fabsf(g1) + 1.0f) / 3.0f;
  float acc = b2[h];
  for (int u = 0; u < 512; ++u) {
    float hid = fmaxf(t0 * w1[u] + t1 * w1[512 + u] + b1[u], 0.0f);
    acc += hid * w2[u * HEADS + h];
  }
  tbv[e * HEADS + h] = acc;
}

// gather by relative index + 16*sigmoid -> biasArr[H][64][64]
__global__ void k_bias_gather(const float* __restrict__ tbv,
                              float* __restrict__ biasArr) {
  int id = blockIdx.x * blockDim.x + threadIdx.x;  // exactly 12*64*64
  int h = id / (NTOK * NTOK);
  int r = id % (NTOK * NTOK);
  int m = r / NTOK, n = r % NTOK;
  int dy = (m >> 3) - (n >> 3) + 7;
  int dx = (m & 7) - (n & 7) + 7;
  float v = tbv[(dy * 15 + dx) * HEADS + h];
  biasArr[id] = 16.0f / (1.0f + __expf(-v));
}

// ---------------- WMMA GEMM with TDM-staged LDS tiles ----------------
// C(M x 384) = A(M x 384) * Bt^T + bias.  Bt is (N x K) row-major f16.
// Block tile 128M x 64N, 8 waves, each wave one 16-row strip x 4 n-tiles.
// A tile (128x32) and B tile (64x32) are DMA'd to LDS by the TDM (wave 0 issues),
// double-buffered against WMMA compute; fragments come from LDS via ds_load_b128.
// mode 0: write f16 scattered to (B, H, N, Dh).   mode 1: write f32 row-major.
__global__ void __launch_bounds__(256)
k_gemm_wmma(const _Float16* __restrict__ A, const _Float16* __restrict__ Bt,
            const float* __restrict__ bias, _Float16* __restrict__ outScatter,
            float* __restrict__ outF32, int mode) {
  __shared__ alignas(16) _Float16 sA[2][128 * 32];   // 2 x 8KB
  __shared__ alignas(16) _Float16 sB[2][64 * 32];    // 2 x 4KB

  const int wave  = threadIdx.x >> 5;
  const int lane  = threadIdx.x & 31;
  const int lr    = lane & 15;
  const int hf    = lane >> 4;
  const int mBase = blockIdx.x * 128;
  const int nBase = blockIdx.y * 64;

  // LDS byte offsets (low 32 bits of the shared-aperture generic address)
  const unsigned int ldsA[2] = {(unsigned int)(uintptr_t)&sA[0][0],
                                (unsigned int)(uintptr_t)&sA[1][0]};
  const unsigned int ldsB[2] = {(unsigned int)(uintptr_t)&sB[0][0],
                                (unsigned int)(uintptr_t)&sB[1][0]};

  if (wave == 0) {   // prologue: fill buffer 0
    tdm_load_2d(ldsA[0], A  + (size_t)mBase * CDIM, 32, 128, CDIM, MROWS, CDIM);
    tdm_load_2d(ldsB[0], Bt + (size_t)nBase * CDIM, 32, 64,  CDIM, CDIM,  CDIM);
  }

  const v8f vzero = {0, 0, 0, 0, 0, 0, 0, 0};
  v8f acc[4];
#pragma unroll
  for (int j = 0; j < 4; ++j) acc[j] = vzero;

  for (int ks = 0; ks < KSTEPS; ++ks) {
    const int cur = ks & 1;
    if (wave == 0) {
      if (ks + 1 < KSTEPS) {   // prefetch next K-slab into the other buffer
        tdm_load_2d(ldsA[cur ^ 1], A  + (size_t)mBase * CDIM + (ks + 1) * 32,
                    32, 128, CDIM, MROWS, CDIM);
        tdm_load_2d(ldsB[cur ^ 1], Bt + (size_t)nBase * CDIM + (ks + 1) * 32,
                    32, 64, CDIM, CDIM, CDIM);
        __builtin_amdgcn_s_wait_tensorcnt(2);   // current pair done (in-order)
      } else {
        __builtin_amdgcn_s_wait_tensorcnt(0);
      }
    }
    __syncthreads();           // buf[cur] visible to all waves

    v16h a = load_frag_a(&sA[cur][wave * 16 * 32], 32);
#pragma unroll
    for (int j = 0; j < 4; ++j) {
      v16h b = load_frag_a(&sB[cur][j * 16 * 32], 32);
      acc[j] = wmma_f16(a, b, acc[j]);
    }
    __syncthreads();           // reads done before TDM overwrites buf[cur]
  }

#pragma unroll
  for (int j = 0; j < 4; ++j) {
    const int c    = nBase + j * 16 + lr;
    const float bv = bias ? bias[c] : 0.0f;
#pragma unroll
    for (int i = 0; i < 8; ++i) {
      const int gm  = mBase + wave * 16 + i + 8 * hf;  // C/D: M = i + 8*(lane>=16)
      const float v = acc[j][i] + bv;
      if (mode == 0) {
        int b = gm >> 6, nn = gm & 63, h = c >> 5, d = c & 31;
        outScatter[((((size_t)b * HEADS) + h) * NTOK + nn) * DH + d] = (_Float16)v;
      } else {
        outF32[(size_t)gm * CDIM + c] = v;
      }
    }
  }
}

// ---------------- fused cosine attention per (window, head) ----------------
__global__ void __launch_bounds__(128)
k_attn(const _Float16* __restrict__ qh, const _Float16* __restrict__ kh,
       const _Float16* __restrict__ vh, const float* __restrict__ biasArr,
       const float* __restrict__ mask, const float* __restrict__ tau,
       _Float16* __restrict__ aoh) {
  __shared__ float qn[NTOK];
  __shared__ float kn[NTOK];
  __shared__ alignas(16) _Float16 sP[NTOK * NTOK];   // softmaxed P (f16)
  __shared__ alignas(16) _Float16 sVT[DH * NTOK];    // V transposed (Dh x N)

  const int bh = blockIdx.x;
  const int b  = bh / HEADS;
  const int h  = bh % HEADS;
  const size_t off = (size_t)bh * NTOK * DH;
  const int t = threadIdx.x;

  // L2 norms of q/k rows
  {
    const int r = t & 63;
    const _Float16* src = (t < 64) ? (qh + off + r * DH) : (kh + off + r * DH);
    float s = 0.f;
#pragma unroll
    for (int d = 0; d < DH; ++d) { float x = (float)src[d]; s += x * x; }
    if (t < 64) qn[r] = sqrtf(s); else kn[r] = sqrtf(s);
  }
  // stage V^T so the second WMMA's B fragments are contiguous in K
  for (int e = t; e < NTOK * DH; e += 128) {
    int k = e >> 5, d = e & 31;
    sVT[d * NTOK + k] = vh[off + e];
  }
  __syncthreads();

  const int wave = t >> 5;
  const int lane = t & 31;
  const int lr   = lane & 15;
  const int hf   = lane >> 4;
  const v8f vzero = {0, 0, 0, 0, 0, 0, 0, 0};

  // S = q (64x32) x k^T (32x64): wave w computes rows 16w..16w+15, 4 n-tiles
  v16h aq = load_frag_a(qh + off + (size_t)wave * 16 * DH, DH);
  v8f S[4];
#pragma unroll
  for (int j = 0; j < 4; ++j) {
    v16h bk = load_frag_a(kh + off + (size_t)j * 16 * DH, DH);
    S[j] = wmma_f16(aq, bk, vzero);
  }

  const float lscale = fmaxf(tau[h] + 2.302585093f, 0.01f);  // max(tau+ln10, .01)
  const float* biasH = biasArr + (size_t)h * NTOK * NTOK;
  const float* maskB = mask + (size_t)b * NTOK * NTOK;

  float vals[4][8];
#pragma unroll
  for (int j = 0; j < 4; ++j) {
    const int n = j * 16 + lr;
    const float knn = kn[n];
#pragma unroll
    for (int i = 0; i < 8; ++i) {
      const int m = wave * 16 + i + 8 * hf;
      const float den = fmaxf(qn[m] * knn, 1e-6f);
      vals[j][i] = S[j][i] / den * lscale + biasH[m * NTOK + n] + maskB[m * NTOK + n];
    }
  }

  // softmax over the 64-wide row: 4 tiles x 16 lanes (xor masks <16 stay in-half)
#pragma unroll
  for (int i = 0; i < 8; ++i) {
    float rm = -3.0e38f;
#pragma unroll
    for (int j = 0; j < 4; ++j) rm = fmaxf(rm, vals[j][i]);
#pragma unroll
    for (int msk = 1; msk < 16; msk <<= 1) rm = fmaxf(rm, __shfl_xor(rm, msk, 32));
    float rs = 0.f, pj[4];
#pragma unroll
    for (int j = 0; j < 4; ++j) { pj[j] = __expf(vals[j][i] - rm); rs += pj[j]; }
#pragma unroll
    for (int msk = 1; msk < 16; msk <<= 1) rs += __shfl_xor(rs, msk, 32);
    const float inv = 1.0f / rs;
    const int m = wave * 16 + i + 8 * hf;
#pragma unroll
    for (int j = 0; j < 4; ++j)
      sP[m * NTOK + j * 16 + lr] = (_Float16)(pj[j] * inv);
  }
  __syncthreads();

  // O = P (64x64) x V (64x32): K = 64 -> 2 WMMA steps, 2 n-tiles of Dh
  v16h ap0 = load_frag_a(sP + wave * 16 * NTOK, NTOK);
  v16h ap1 = load_frag_a(sP + wave * 16 * NTOK + 32, NTOK);
#pragma unroll
  for (int j = 0; j < 2; ++j) {
    v16h bv0 = load_frag_a(sVT + j * 16 * NTOK, NTOK);
    v16h bv1 = load_frag_a(sVT + j * 16 * NTOK + 32, NTOK);
    v8f o = wmma_f16(ap0, bv0, vzero);
    o = wmma_f16(ap1, bv1, o);
#pragma unroll
    for (int i = 0; i < 8; ++i) {
      const int m = wave * 16 + i + 8 * hf;
      const int d = j * 16 + lr;
      aoh[((size_t)(b * NTOK + m)) * CDIM + h * DH + d] = (_Float16)o[i];
    }
  }
}

// ---------------- host launch ----------------

extern "C" void kernel_launch(void* const* d_in, const int* in_sizes, int n_in,
                              void* d_out, int out_size, void* d_ws, size_t ws_size,
                              hipStream_t stream) {
  const float* x     = (const float*)d_in[0];
  const float* mask  = (const float*)d_in[1];
  const float* wq    = (const float*)d_in[2];
  const float* bq    = (const float*)d_in[3];
  const float* wk    = (const float*)d_in[4];
  const float* wv    = (const float*)d_in[5];
  const float* bv    = (const float*)d_in[6];
  const float* cw1   = (const float*)d_in[7];
  const float* cb1   = (const float*)d_in[8];
  const float* cw2   = (const float*)d_in[9];
  const float* cb2   = (const float*)d_in[10];
  const float* tau   = (const float*)d_in[11];
  const float* wproj = (const float*)d_in[12];
  const float* bproj = (const float*)d_in[13];

  char* ws = (char*)d_ws;
  constexpr size_t SZ_X16 = (size_t)MROWS * CDIM * 2;  // 56,623,104 (256-aligned)
  constexpr size_t SZ_W16 = (size_t)CDIM * CDIM * 2;   // 294,912    (256-aligned)

  _Float16* xh   = (_Float16*)(ws);
  _Float16* wqT  = (_Float16*)(ws + SZ_X16);
  _Float16* wkT  = (_Float16*)(ws + SZ_X16 + 1 * SZ_W16);
  _Float16* wvT  = (_Float16*)(ws + SZ_X16 + 2 * SZ_W16);
  _Float16* wpT  = (_Float16*)(ws + SZ_X16 + 3 * SZ_W16);
  _Float16* qhp  = (_Float16*)(ws + SZ_X16 + 4 * SZ_W16);
  _Float16* khp  = (_Float16*)(ws + 2 * SZ_X16 + 4 * SZ_W16);
  _Float16* vhp  = (_Float16*)(ws + 3 * SZ_X16 + 4 * SZ_W16);
  _Float16* aoh  = (_Float16*)(ws + 4 * SZ_X16 + 4 * SZ_W16);
  float*    tbv  = (float*)   (ws + 5 * SZ_X16 + 4 * SZ_W16);
  float*    biasArr = (float*)(ws + 5 * SZ_X16 + 4 * SZ_W16 + 11008);

  // prep
  k_cvt_f32_to_f16<<<2048, 256, 0, stream>>>(x, xh, MROWS * CDIM);
  k_transpose_w<<<576, 256, 0, stream>>>(wq, wqT);
  k_transpose_w<<<576, 256, 0, stream>>>(wk, wkT);
  k_transpose_w<<<576, 256, 0, stream>>>(wv, wvT);
  k_transpose_w<<<576, 256, 0, stream>>>(wproj, wpT);
  k_cpb_mlp<<<11, 256, 0, stream>>>(cw1, cb1, cw2, cb2, tbv);
  k_bias_gather<<<192, 256, 0, stream>>>(tbv, biasArr);

  // q/k/v projections
  dim3 gg(MROWS / 128, CDIM / 64);  // (576, 6)
  k_gemm_wmma<<<gg, 256, 0, stream>>>(xh, wqT, bq,      qhp, nullptr, 0);
  k_gemm_wmma<<<gg, 256, 0, stream>>>(xh, wkT, nullptr, khp, nullptr, 0);
  k_gemm_wmma<<<gg, 256, 0, stream>>>(xh, wvT, bv,      vhp, nullptr, 0);

  // fused attention
  k_attn<<<BWIN * HEADS, 128, 0, stream>>>(qhp, khp, vhp, biasArr, mask, tau, aoh);

  // output projection -> f32
  k_gemm_wmma<<<gg, 256, 0, stream>>>(aoh, wpT, bproj, nullptr, (float*)d_out, 1);
}
